// PFPEncoder_11012296147740
// MI455X (gfx1250) — matrix operand order repaired and checked by reading, compiled, not run on hardware
//
#include <hip/hip_runtime.h>
#include <hip/hip_bf16.h>

// ---------------------------------------------------------------------------
// PFP encoder (NNConv message passing) for MI455X / gfx1250, wave32 + WMMA +
// async global->LDS staging of the shared B operand.
//
#define N_NODES  10000
#define N_EDGES  30000
#define NODE_DIM 75
#define BOND_DIM 16
#define IN_DIM   64
#define INTERNAL 128
#define FP_DIM   4096
#define N_GRAPHS 16
#define DEPTH    3

typedef __attribute__((ext_vector_type(16))) __bf16 v16bf;
typedef __attribute__((ext_vector_type(8)))  float  v8f;
typedef __attribute__((ext_vector_type(4)))  unsigned int u32x4;

union Frag16 {            // one WMMA 16-bit A/B fragment (8 VGPRs)
  v16bf  v;
  __bf16 e[16];
  u32x4  q[2];
};

// A-matrix 16x32 (16-bit) lane/element -> K mapping (contiguous, matches packed B).
// lanes 0-15: elements 0..15 -> K 0..15 ; lanes 16-31: K 16..31.
__device__ __forceinline__ int kmapA(int lh, int t) {
  return 16 * lh + t;
}

// Byte offset of a generic pointer within the workgroup LDS allocation
// (addrspacecast generic -> AS(3); AS(3) pointers are 32-bit LDS offsets).
__device__ __forceinline__ unsigned lds_offset(const void* p) {
  return (unsigned)(unsigned long long)(const __attribute__((address_space(3))) char*)p;
}

// ---------------------------------------------------------------------------
// msg0 = relu(x @ W_ae + b_ae)   [N, 64]  (fp32 + bf16 copies)
__global__ __launch_bounds__(256)
void k_node_init(const float* __restrict__ x, const float* __restrict__ W_ae,
                 const float* __restrict__ b_ae,
                 float* __restrict__ msg, __bf16* __restrict__ msg_bf) {
  int gid = blockIdx.x * blockDim.x + threadIdx.x;
  if (gid >= N_NODES * IN_DIM) return;
  int n = gid / IN_DIM, o = gid % IN_DIM;
  float acc = b_ae[o];
  const float* xr = x + (size_t)n * NODE_DIM;
  for (int d = 0; d < NODE_DIM; ++d) acc += xr[d] * W_ae[d * IN_DIM + o];
  acc = fmaxf(acc, 0.0f);
  msg[gid]    = acc;
  msg_bf[gid] = (__bf16)acc;
}

// Bterm = msg @ reshape(b_e2, [64,64])   [N, 64]  (bilinear bias term)
__global__ __launch_bounds__(256)
void k_bterm(const float* __restrict__ msg, const float* __restrict__ b_e2,
             float* __restrict__ Bterm) {
  int gid = blockIdx.x * blockDim.x + threadIdx.x;
  if (gid >= N_NODES * IN_DIM) return;
  int n = gid / IN_DIM, o = gid % IN_DIM;
  const float* mr = msg + (size_t)n * IN_DIM;
  float acc = 0.0f;
  for (int i = 0; i < IN_DIM; ++i) acc += mr[i] * b_e2[i * IN_DIM + o];
  Bterm[gid] = acc;
}

// h = relu(edge_attr @ W_e1 + b_e1)  [E,128] bf16 ; edge flags + masked counts
__global__ __launch_bounds__(256)
void k_edge_prep(const float* __restrict__ edge_attr,
                 const float* __restrict__ W_e1, const float* __restrict__ b_e1,
                 const int* __restrict__ dst,
                 __bf16* __restrict__ h_bf, int* __restrict__ flags,
                 float* __restrict__ cnt_cov, float* __restrict__ cnt_spa) {
  int gid = blockIdx.x * blockDim.x + threadIdx.x;
  if (gid >= N_EDGES * INTERNAL) return;
  int e = gid / INTERNAL, k = gid % INTERNAL;
  const float* ar = edge_attr + (size_t)e * BOND_DIM;
  float acc = b_e1[k];
  for (int d = 0; d < BOND_DIM; ++d) acc += ar[d] * W_e1[d * INTERNAL + k];
  h_bf[gid] = (__bf16)fmaxf(acc, 0.0f);
  if (k == 0) {
    int fl = (ar[BOND_DIM - 1] != 0.0f) ? 1 : 0;   // 1 = spatial
    flags[e] = fl;
    atomicAdd(fl ? &cnt_spa[dst[e]] : &cnt_cov[dst[e]], 1.0f);
  }
}

// Pre-swizzle W_e2 into per-lane WMMA B-fragment order:
// W2sw[((c*4+nt)*32+lane)*16 + t] = bf16( W_e2[k, i*64+o] ),
//   j = c*32 + 16*(lane>>4) + t ; k = j>>6 ; i = j&63 ; o = nt*16 + (lane&15)
__global__ __launch_bounds__(256)
void k_w2_pack(const float* __restrict__ W_e2, __bf16* __restrict__ W2sw) {
  int gid = blockIdx.x * blockDim.x + threadIdx.x;
  if (gid >= 256 * 4 * 32 * 16) return;
  int t = gid & 15, lane = (gid >> 4) & 31, nt = (gid >> 9) & 3, c = gid >> 11;
  int lh = lane >> 4;
  int j = c * 32 + 16 * lh + t;
  int k = j >> 6, i = j & 63;
  int o = nt * 16 + (lane & 15);
  W2sw[gid] = (__bf16)W_e2[(size_t)k * (IN_DIM * IN_DIM) + i * IN_DIM + o];
}

// Pre-swizzle W_fp [64,4096] the same way (2 K-chunks x 256 N-tiles).
__global__ __launch_bounds__(256)
void k_wfp_pack(const float* __restrict__ W_fp, __bf16* __restrict__ Wfpsw) {
  int gid = blockIdx.x * blockDim.x + threadIdx.x;
  if (gid >= 2 * 256 * 32 * 16) return;
  int t = gid & 15, lane = (gid >> 4) & 31, ntg = (gid >> 9) & 255, kc = gid >> 17;
  int lh = lane >> 4;
  int j = kc * 32 + 16 * lh + t;                       // 0..63 input dim
  int o = ntg * 16 + (lane & 15);
  Wfpsw[gid] = (__bf16)W_fp[(size_t)j * FP_DIM + o];
}

// ---------------------------------------------------------------------------
// Fused bilinear NNConv edge transform + masked scatter aggregation.
// per_edge[16 edges, 64] = (h x m_src)[16, 8192] @ W2'[8192, 64]  via WMMA bf16.
// K chunked by 32 (within a chunk k is constant => A_frag = h[e,k] * m_frag).
// All 8 waves of the block share the same B stream: staged in LDS with
// double-buffered global_load_async_to_lds_b128 (ASYNCcnt-tracked).
#define CPP     4                         // K-chunks per staging phase
#define PHASES  (256 / CPP)               // 64 phases
#define CHUNK_B 4096                      // bytes per 32-wide K chunk of B
__global__ __launch_bounds__(256)
void k_bilinear(const __bf16* __restrict__ msg_bf, const __bf16* __restrict__ h_bf,
                const __bf16* __restrict__ W2sw, const float* __restrict__ Bterm,
                const int* __restrict__ src, const int* __restrict__ dst,
                const int* __restrict__ flags,
                float* __restrict__ cov_s, float* __restrict__ spa_s, int n_tiles) {
  __shared__ __align__(16) __bf16 sB[2][CPP * (CHUNK_B / 2)];   // 2 x 16 KB

  int tid  = threadIdx.x;
  int lane = tid & 31;
  int lh = lane >> 4, lm = lane & 15;
  int wid = blockIdx.x * 8 + (tid >> 5);            // wave-uniform
  bool active = wid < n_tiles;
  int tile = active ? wid : (n_tiles - 1);          // clamp: keep EXEC full
  int e0 = tile * 16;

  // Two m-fragments (i in [0,32) and [32,64)) for this lane's edge row.
  int my_edge = e0 + lm;
  const __bf16* mrow = msg_bf + (size_t)src[my_edge] * IN_DIM;
  Frag16 mf0, mf1;
#pragma unroll
  for (int t = 0; t < 16; ++t) {
    int K = kmapA(lh, t);
    mf0.e[t] = mrow[K];
    mf1.e[t] = mrow[32 + K];
  }
  const __bf16* hrow = h_bf + (size_t)my_edge * INTERNAL;

  const char* gB = (const char*)W2sw;
  unsigned sB0 = lds_offset(&sB[0][0]);
  unsigned sB1 = lds_offset(&sB[1][0]);

  // Issue phase 0 (each thread: one b128 per chunk; 256*16B = one 4KB chunk).
  {
    const char* g = gB + (size_t)tid * 16;
#pragma unroll
    for (int q = 0; q < CPP; ++q) {
      unsigned d = sB0 + q * CHUNK_B + tid * 16;
      const char* ga = g + q * CHUNK_B;
      asm volatile("global_load_async_to_lds_b128 %0, %1, off"
                   :: "v"(d), "v"(ga) : "memory");
    }
  }

  v8f acc0 = {}, acc1 = {}, acc2 = {}, acc3 = {};
  for (int p = 0; p < PHASES; ++p) {
    if (p + 1 < PHASES) {                           // prefetch next phase
      unsigned dbase = ((p + 1) & 1) ? sB1 : sB0;
      const char* g = gB + (size_t)(p + 1) * (CPP * CHUNK_B) + (size_t)tid * 16;
#pragma unroll
      for (int q = 0; q < CPP; ++q) {
        unsigned d = dbase + q * CHUNK_B + tid * 16;
        const char* ga = g + q * CHUNK_B;
        asm volatile("global_load_async_to_lds_b128 %0, %1, off"
                     :: "v"(d), "v"(ga) : "memory");
      }
      // in-order completion: <=4 outstanding => this phase's 4 have landed
      asm volatile("s_wait_asynccnt 0x4" ::: "memory");
    } else {
      asm volatile("s_wait_asynccnt 0x0" ::: "memory");
    }
    __syncthreads();                                // all waves' parts visible

    const __bf16* buf = &sB[p & 1][0];
#pragma unroll
    for (int q = 0; q < CPP; ++q) {
      int c = p * CPP + q;
      float hv = (float)hrow[c >> 1];
      const Frag16& m = (c & 1) ? mf1 : mf0;
      Frag16 a;
#pragma unroll
      for (int t = 0; t < 16; ++t) a.e[t] = (__bf16)((float)m.e[t] * hv);

      const u32x4* pb = (const u32x4*)(buf + q * (CHUNK_B / 2)) + lane * 2;
      Frag16 b0, b1, b2, b3;                        // nt stride = 32 lanes * 2
      b0.q[0] = pb[0];       b0.q[1] = pb[1];
      b1.q[0] = pb[0 + 64];  b1.q[1] = pb[1 + 64];
      b2.q[0] = pb[0 + 128]; b2.q[1] = pb[1 + 128];
      b3.q[0] = pb[0 + 192]; b3.q[1] = pb[1 + 192];

      acc0 = __builtin_amdgcn_wmma_f32_16x16x32_bf16(false, a.v, false, b0.v, (short)0, acc0, false, false);
      acc1 = __builtin_amdgcn_wmma_f32_16x16x32_bf16(false, a.v, false, b1.v, (short)0, acc1, false, false);
      acc2 = __builtin_amdgcn_wmma_f32_16x16x32_bf16(false, a.v, false, b2.v, (short)0, acc2, false, false);
      acc3 = __builtin_amdgcn_wmma_f32_16x16x32_bf16(false, a.v, false, b3.v, (short)0, acc3, false, false);
    }
    __syncthreads();                                // buffer reuse fence
  }

  if (active) {
    // Epilogue: + Bterm[src], masked scatter-mean numerators via atomics.
#pragma unroll
    for (int r = 0; r < 8; ++r) {
      int e  = e0 + r + 8 * lh;                     // C/D layout: M = r + 8*lh
      int se = src[e], de = dst[e];
      float* tgt = flags[e] ? spa_s : cov_s;
      const float* bt = Bterm + (size_t)se * IN_DIM;
      float* trow = tgt + (size_t)de * IN_DIM;
      atomicAdd(&trow[0 * 16 + lm], acc0[r] + bt[0 * 16 + lm]);
      atomicAdd(&trow[1 * 16 + lm], acc1[r] + bt[1 * 16 + lm]);
      atomicAdd(&trow[2 * 16 + lm], acc2[r] + bt[2 * 16 + lm]);
      atomicAdd(&trow[3 * 16 + lm], acc3[r] + bt[3 * 16 + lm]);
    }
  }
}

// msg_new = relu(cov_mean + spa_mean + 2*(msg @ W_root + b_conv))
__global__ __launch_bounds__(256)
void k_node_update(const float* __restrict__ msg_old, const float* __restrict__ W_root,
                   const float* __restrict__ b_conv,
                   const float* __restrict__ cov_s, const float* __restrict__ spa_s,
                   const float* __restrict__ cnt_cov, const float* __restrict__ cnt_spa,
                   float* __restrict__ msg_new, __bf16* __restrict__ msg_bf) {
  int gid = blockIdx.x * blockDim.x + threadIdx.x;
  if (gid >= N_NODES * IN_DIM) return;
  int n = gid / IN_DIM, o = gid % IN_DIM;
  const float* mr = msg_old + (size_t)n * IN_DIM;
  float root = b_conv[o];
  for (int i = 0; i < IN_DIM; ++i) root += mr[i] * W_root[i * IN_DIM + o];
  float cc = cnt_cov[n], cs = cnt_spa[n];
  float cv = (cc > 0.0f) ? cov_s[gid] / fmaxf(cc, 1.0f) : 0.0f;
  float sv = (cs > 0.0f) ? spa_s[gid] / fmaxf(cs, 1.0f) : 0.0f;
  float v = fmaxf(cv + sv + 2.0f * root, 0.0f);
  msg_new[gid] = v;
  msg_bf[gid]  = (__bf16)v;
}

// expanded = relu(msg @ W_fp + b_fp); fingerprint += segment_sum(expanded, batch)
// One wave: 16 nodes x 64 fp-columns (4 WMMA N-tiles, K = 64 in 2 chunks).
__global__ __launch_bounds__(256)
void k_fp_expand(const __bf16* __restrict__ msg_bf, const __bf16* __restrict__ Wfpsw,
                 const float* __restrict__ b_fp, const int* __restrict__ batch,
                 float* __restrict__ out) {
  int wid = blockIdx.x * 8 + (threadIdx.x >> 5);    // wave-uniform
  if (wid >= (N_NODES / 16) * (FP_DIM / 64)) return;
  int mtile = wid >> 6, nq = wid & 63;
  int lane = threadIdx.x & 31;
  int lh = lane >> 4, lm = lane & 15;

  const __bf16* mrow = msg_bf + (size_t)(mtile * 16 + lm) * IN_DIM;
  Frag16 a0, a1;
#pragma unroll
  for (int t = 0; t < 16; ++t) {
    int K = kmapA(lh, t);
    a0.e[t] = mrow[K];
    a1.e[t] = mrow[32 + K];
  }

  v8f acc0 = {}, acc1 = {}, acc2 = {}, acc3 = {};
  const u32x4* wb = (const u32x4*)Wfpsw;
#pragma unroll
  for (int kc = 0; kc < 2; ++kc) {
    const v16bf& av = kc ? a1.v : a0.v;
    const u32x4* p = wb + (((size_t)kc * 256 + nq * 4) * 32 + lane) * 2;
    Frag16 b0, b1, b2, b3;
    b0.q[0] = p[0];       b0.q[1] = p[1];
    b1.q[0] = p[0 + 64];  b1.q[1] = p[1 + 64];
    b2.q[0] = p[0 + 128]; b2.q[1] = p[1 + 128];
    b3.q[0] = p[0 + 192]; b3.q[1] = p[1 + 192];
    acc0 = __builtin_amdgcn_wmma_f32_16x16x32_bf16(false, av, false, b0.v, (short)0, acc0, false, false);
    acc1 = __builtin_amdgcn_wmma_f32_16x16x32_bf16(false, av, false, b1.v, (short)0, acc1, false, false);
    acc2 = __builtin_amdgcn_wmma_f32_16x16x32_bf16(false, av, false, b2.v, (short)0, acc2, false, false);
    acc3 = __builtin_amdgcn_wmma_f32_16x16x32_bf16(false, av, false, b3.v, (short)0, acc3, false, false);
  }

  int colb = nq * 64 + lm;
#pragma unroll
  for (int r = 0; r < 8; ++r) {
    int node = mtile * 16 + r + 8 * lh;
    float* orow = out + (size_t)batch[node] * FP_DIM;
    atomicAdd(&orow[colb +  0], fmaxf(acc0[r] + b_fp[colb +  0], 0.0f));
    atomicAdd(&orow[colb + 16], fmaxf(acc1[r] + b_fp[colb + 16], 0.0f));
    atomicAdd(&orow[colb + 32], fmaxf(acc2[r] + b_fp[colb + 32], 0.0f));
    atomicAdd(&orow[colb + 48], fmaxf(acc3[r] + b_fp[colb + 48], 0.0f));
  }
}

// ---------------------------------------------------------------------------
static inline size_t alignUp(size_t x) { return (x + 255) & ~(size_t)255; }

extern "C" void kernel_launch(void* const* d_in, const int* in_sizes, int n_in,
                              void* d_out, int out_size, void* d_ws, size_t ws_size,
                              hipStream_t stream) {
  const float* x         = (const float*)d_in[0];
  const int*   edge_idx  = (const int*)d_in[1];
  const float* edge_attr = (const float*)d_in[2];
  const int*   batch     = (const int*)d_in[3];
  const float* W_ae      = (const float*)d_in[4];
  const float* b_ae      = (const float*)d_in[5];
  const float* W_e1      = (const float*)d_in[6];
  const float* b_e1      = (const float*)d_in[7];
  const float* W_e2      = (const float*)d_in[8];
  const float* b_e2      = (const float*)d_in[9];
  const float* W_root    = (const float*)d_in[10];
  const float* b_conv    = (const float*)d_in[11];
  const float* W_fp      = (const float*)d_in[12];
  const float* b_fp      = (const float*)d_in[13];
  const int* src = edge_idx;
  const int* dst = edge_idx + N_EDGES;

  // Carve workspace.
  char* w = (char*)d_ws;
  size_t off = 0;
  auto carve = [&](size_t bytes) { void* p = w + off; off = alignUp(off + bytes); return p; };
  float*  msgA    = (float*) carve((size_t)N_NODES * IN_DIM * 4);
  float*  msgB    = (float*) carve((size_t)N_NODES * IN_DIM * 4);
  __bf16* msg_bf  = (__bf16*)carve((size_t)N_NODES * IN_DIM * 2);
  float*  Bterm   = (float*) carve((size_t)N_NODES * IN_DIM * 4);
  __bf16* h_bf    = (__bf16*)carve((size_t)N_EDGES * INTERNAL * 2);
  int*    flags   = (int*)   carve((size_t)N_EDGES * 4);
  __bf16* W2sw    = (__bf16*)carve((size_t)256 * 4 * 32 * 16 * 2);
  __bf16* Wfpsw   = (__bf16*)carve((size_t)2 * 256 * 32 * 16 * 2);
  float*  cov_s   = (float*) carve((size_t)N_NODES * IN_DIM * 4);
  float*  spa_s   = (float*) carve((size_t)N_NODES * IN_DIM * 4);
  float*  cnt_cov = (float*) carve((size_t)N_NODES * 4);
  float*  cnt_spa = (float*) carve((size_t)N_NODES * 4);
  (void)ws_size; (void)n_in; (void)in_sizes; (void)out_size;

  // One-time (per call) zeroing + prep.
  hipMemsetAsync(d_out, 0, (size_t)N_GRAPHS * FP_DIM * 4, stream);
  hipMemsetAsync(cnt_cov, 0, (size_t)N_NODES * 4, stream);
  hipMemsetAsync(cnt_spa, 0, (size_t)N_NODES * 4, stream);

  k_node_init<<<(N_NODES * IN_DIM + 255) / 256, 256, 0, stream>>>(x, W_ae, b_ae, msgA, msg_bf);
  k_bterm<<<(N_NODES * IN_DIM + 255) / 256, 256, 0, stream>>>(msgA, b_e2, Bterm);
  k_edge_prep<<<(N_EDGES * INTERNAL + 255) / 256, 256, 0, stream>>>(
      edge_attr, W_e1, b_e1, dst, h_bf, flags, cnt_cov, cnt_spa);
  k_w2_pack<<<(256 * 4 * 32 * 16) / 256, 256, 0, stream>>>(W_e2, W2sw);
  k_wfp_pack<<<(2 * 256 * 32 * 16) / 256, 256, 0, stream>>>(W_fp, Wfpsw);

  const int n_tiles = N_EDGES / 16;                 // 1875
  const int bil_blocks = (n_tiles + 7) / 8;         // 8 waves / block
  const int fp_waves = (N_NODES / 16) * (FP_DIM / 64);
  const int fp_blocks = (fp_waves + 7) / 8;

  float* msg_cur = msgA;
  float* msg_nxt = msgB;
  for (int d = 0; d < DEPTH; ++d) {
    hipMemsetAsync(cov_s, 0, (size_t)N_NODES * IN_DIM * 4, stream);
    hipMemsetAsync(spa_s, 0, (size_t)N_NODES * IN_DIM * 4, stream);
    k_bilinear<<<bil_blocks, 256, 0, stream>>>(
        msg_bf, h_bf, W2sw, Bterm, src, dst, flags, cov_s, spa_s, n_tiles);
    k_node_update<<<(N_NODES * IN_DIM + 255) / 256, 256, 0, stream>>>(
        msg_cur, W_root, b_conv, cov_s, spa_s, cnt_cov, cnt_spa, msg_nxt, msg_bf);
    k_bterm<<<(N_NODES * IN_DIM + 255) / 256, 256, 0, stream>>>(msg_nxt, b_e2, Bterm);
    k_fp_expand<<<fp_blocks, 256, 0, stream>>>(msg_bf, Wfpsw, b_fp, batch, (float*)d_out);
    float* tmp = msg_cur; msg_cur = msg_nxt; msg_nxt = tmp;
  }
}
